// GRU_BAYES_47210280518311
// MI455X (gfx1250) — compile-verified, matrix operand order checked by reading, not run on hardware
//
#include <hip/hip_runtime.h>
#include <hip/hip_bf16.h>

// ATAE-style GRU + attention head for MI455X (gfx1250, wave32, WMMA, TDM).
//   k_aspool   : aspect mean-pool (B x E), f32
//   k_cvtT     : f32 [K][N] -> f16 [N][K] transpose-convert (w_ih, w_hh)
//   k_xin      : xin = [aspect | masked embed] f16 (32768 x 1024)
//   k_gemm_xg  : xg = xin @ w_ih + b via v_wmma_f32_16x16x32_f16,
//                A/B tiles staged by TENSOR_LOAD_TO_LDS (double buffered)
//   k_gru      : 8 persistent blocks x 16 batch rows; h@w_hh via WMMA with
//                B fragments read directly from L2-resident transposed w_hh
//   k_qv       : v = wk_w @ ((q@wq_w+wq_b)@bil_w); aspect part+wk_b cancel
//   k_logits   : logit[b,t] = h[b,t].v (wave-per-row dot)
//   k_softpool : softmax over T, pooled = score@h, out = pooled@dense_w + b

typedef _Float16 half_t;
typedef __attribute__((ext_vector_type(16))) _Float16 v16h;
typedef __attribute__((ext_vector_type(8)))  float    v8f;
typedef __attribute__((ext_vector_type(4)))  unsigned int v4u;
typedef __attribute__((ext_vector_type(8)))  int v8i;
typedef __attribute__((ext_vector_type(4)))  int v4i;

#define E_DIM 512
#define H_DIM 512
#define POL_N 3
#define B_DIM 128
#define T_DIM 256
#define TA_DIM 8
#define D_DIM 1024
#define I_DIM 1024   // 2E (also row length K of transposed w_ih)
#define G_DIM 1536   // 3H
#define M_DIM (B_DIM * T_DIM)  // 32768

#if defined(__gfx1250__) && __has_builtin(__builtin_amdgcn_tensor_load_to_lds) && \
    __has_builtin(__builtin_amdgcn_s_wait_tensorcnt)
#define USE_TDM 1
#else
#define USE_TDM 0
#endif

__device__ __forceinline__ v8f vzero8() {
  v8f z;
#pragma unroll
  for (int i = 0; i < 8; ++i) z[i] = 0.0f;
  return z;
}

__device__ __forceinline__ v8f wmma_f16(v16h a, v16h b, v8f c) {
  return __builtin_amdgcn_wmma_f32_16x16x32_f16(false, a, false, b, (short)0, c,
                                                false, false);
}

// One 16(row)x32(K) f16 fragment from row-major [16][stride] memory (LDS or
// global). CDNA5 ISA 7.12.2: row = lane&15; K halves {0..7,16..23} (lanes
// 0-15) / {8..15,24..31} (lanes 16-31) -> two contiguous 16B loads per lane.
__device__ __forceinline__ v16h load_frag(const half_t* base, int stride, int lane) {
  const half_t* p = base + (lane & 15) * stride + ((lane & 16) ? 8 : 0);
  v16h f;
#pragma unroll
  for (int i = 0; i < 8; ++i) f[i] = p[i];
#pragma unroll
  for (int i = 0; i < 8; ++i) f[8 + i] = p[16 + i];
  return f;
}

#if USE_TDM
// Issue a 2-D TDM load of a [128 rows x 32 halfs] tile (rows strided by
// stride_elems f16 elements) into LDS at byte offset lds_off.
// D# layout per CDNA5 ISA.
// This toolchain's builtin is the 6-arg form:
//   (uint32x4 g0, int32x8 g1, int32x4 g2, int32x4 g3, int32x8, i32 cpol)
__device__ __forceinline__ void tdm_load_tile(const half_t* g, unsigned lds_off,
                                              unsigned stride_elems) {
  unsigned long long ga = (unsigned long long)(uintptr_t)g;
  v4u g0;
  g0[0] = 1u;                                   // count=1 (valid user D#)
  g0[1] = lds_off;                              // lds_addr
  g0[2] = (unsigned)ga;                         // global_addr[31:0]
  g0[3] = (unsigned)(ga >> 32) | (2u << 30);    // global_addr[56:32] | type=2
  v8i g1;
  g1[0] = (int)(1u << 16);                      // data_size=1 -> 2 bytes
  g1[1] = (int)((stride_elems & 0xFFFFu) << 16);            // tensor_dim0 lo16
  g1[2] = (int)((stride_elems >> 16) | 0x80000000u);        // dim0 hi | dim1 lo=32768
  g1[3] = (int)(32u << 16);                     // dim1 hi=0 | tile_dim0=32
  g1[4] = (int)128u;                            // tile_dim1=128, tile_dim2=0
  g1[5] = (int)stride_elems;                    // tensor_dim0_stride lo32
  g1[6] = 0;                                    // stride0 hi, stride1 lo
  g1[7] = 0;                                    // stride1 hi
  v4i z4 = {0, 0, 0, 0};
  v8i z8;
#pragma unroll
  for (int i = 0; i < 8; ++i) z8[i] = 0;
  __builtin_amdgcn_tensor_load_to_lds(g0, g1, z4, z4, z8, 0);
}
#endif

// ---------------------------------------------------------------- aspect pool
__global__ __launch_bounds__(256) void k_aspool(const int* __restrict__ aidx,
                                                const float* __restrict__ emb,
                                                float* __restrict__ asp) {
  int b = blockIdx.x;
  int e = threadIdx.x;
  float s0 = 0.f, s1 = 0.f, cnt = 0.f;
#pragma unroll
  for (int a = 0; a < TA_DIM; ++a) {
    int id = aidx[b * TA_DIM + a];
    if (id != 0) {
      cnt += 1.f;
      s0 += emb[(size_t)id * E_DIM + e];
      s1 += emb[(size_t)id * E_DIM + e + 256];
    }
  }
  float inv = 1.f / cnt;
  asp[b * E_DIM + e] = s0 * inv;
  asp[b * E_DIM + e + 256] = s1 * inv;
}

// --------------------------------- transpose + f32->f16: src[K][N] -> dst[N][K]
__global__ __launch_bounds__(256) void k_cvtT(const float* __restrict__ src,
                                              half_t* __restrict__ dst, int K,
                                              int N) {
  __shared__ float tile[32][33];
  int ktiles = K >> 5;
  int kt = (blockIdx.x % ktiles) << 5;
  int nt = (blockIdx.x / ktiles) << 5;
  int r = threadIdx.x >> 3;          // 0..31
  int c4 = (threadIdx.x & 7) << 2;   // 0,4,..,28
#pragma unroll
  for (int j = 0; j < 4; ++j)
    tile[r][c4 + j] = src[(size_t)(kt + r) * N + nt + c4 + j];
  __syncthreads();
#pragma unroll
  for (int j = 0; j < 4; ++j)
    dst[(size_t)(nt + r) * K + kt + c4 + j] = (half_t)tile[c4 + j][r];
}

// ------------------------------------------------------------------ build xin
__global__ __launch_bounds__(256) void k_xin(const int* __restrict__ tidx,
                                             const float* __restrict__ emb,
                                             const float* __restrict__ asp,
                                             half_t* __restrict__ xin) {
  int bt = blockIdx.x;
  int b = bt / T_DIM;
  int id = tidx[bt];
  float m = (id != 0) ? 1.f : 0.f;
  half_t* row = xin + (size_t)bt * I_DIM;
  int c = threadIdx.x;
  row[c] = (half_t)asp[b * E_DIM + c];
  row[c + 256] = (half_t)asp[b * E_DIM + c + 256];
  row[512 + c] = (half_t)(emb[(size_t)id * E_DIM + c] * m);
  row[512 + c + 256] = (half_t)(emb[(size_t)id * E_DIM + c + 256] * m);
}

// ------------------------------------------------- xg = xin @ w_ih + b (WMMA)
// M=32768, K=1024, N=1536. Block tile 128x128; 8 waves in 4(m) x 2(n) grid,
// each wave 2x4 sixteen-square subtiles. A and B^T tiles staged by TDM with
// ping-pong double buffering (fallback: vectorized copies).
__global__ __launch_bounds__(256) void k_gemm_xg(const half_t* __restrict__ A,
                                                 const half_t* __restrict__ BwT,
                                                 const float* __restrict__ bias,
                                                 half_t* __restrict__ Cout) {
  __shared__ half_t As[2 * 128 * 32];  // 16 KB (ping/pong)
  __shared__ half_t Bs[2 * 128 * 32];  // 16 KB (ping/pong)
  int bm = blockIdx.x & 255, bn = blockIdx.x >> 8;
  int m0 = bm * 128, n0 = bn * 128;
  int tid = threadIdx.x, lane = tid & 31, wave = tid >> 5;
  int mt = (wave & 3) * 32;   // 2 m-tiles
  int nt = (wave >> 2) * 64;  // 4 n-tiles

  v8f acc[2][4];
#pragma unroll
  for (int i = 0; i < 2; ++i)
#pragma unroll
    for (int j = 0; j < 4; ++j) acc[i][j] = vzero8();

#if USE_TDM
  unsigned asOff = (unsigned)(uintptr_t)&As[0];
  unsigned bsOff = (unsigned)(uintptr_t)&Bs[0];
  if (wave == 0) {  // preload stage 0
    tdm_load_tile(A + (size_t)m0 * I_DIM, asOff, I_DIM);
    tdm_load_tile(BwT + (size_t)n0 * I_DIM, bsOff, I_DIM);
  }
#endif

  for (int ks = 0; ks < 32; ++ks) {
#if USE_TDM
    int buf = ks & 1, nbuf = buf ^ 1;
    if (wave == 0) {
      if (ks + 1 < 32) {
        int kk = (ks + 1) * 32;
        tdm_load_tile(A + (size_t)m0 * I_DIM + kk, asOff + nbuf * 8192, I_DIM);
        tdm_load_tile(BwT + (size_t)n0 * I_DIM + kk, bsOff + nbuf * 8192, I_DIM);
        __builtin_amdgcn_s_wait_tensorcnt(2);  // current stage landed
      } else {
        __builtin_amdgcn_s_wait_tensorcnt(0);
      }
    }
    __syncthreads();
    const half_t* as = &As[buf * 4096];
    const half_t* bs = &Bs[buf * 4096];
#else
    {  // stage A and B^T tiles (row-major [row][32]) with 16B copies
      int row = tid >> 1, seg = (tid & 1) * 16;
      const uint4* sa =
          reinterpret_cast<const uint4*>(A + (size_t)(m0 + row) * I_DIM + ks * 32 + seg);
      uint4* da = reinterpret_cast<uint4*>(&As[row * 32 + seg]);
      da[0] = sa[0];
      da[1] = sa[1];
      const uint4* sb =
          reinterpret_cast<const uint4*>(BwT + (size_t)(n0 + row) * I_DIM + ks * 32 + seg);
      uint4* db = reinterpret_cast<uint4*>(&Bs[row * 32 + seg]);
      db[0] = sb[0];
      db[1] = sb[1];
    }
    __syncthreads();
    const half_t* as = As;
    const half_t* bs = Bs;
#endif
    v16h a0 = load_frag(as + (mt + 0) * 32, 32, lane);
    v16h a1 = load_frag(as + (mt + 16) * 32, 32, lane);
#pragma unroll
    for (int j = 0; j < 4; ++j) {
      v16h bf = load_frag(bs + (nt + j * 16) * 32, 32, lane);
      acc[0][j] = wmma_f16(a0, bf, acc[0][j]);
      acc[1][j] = wmma_f16(a1, bf, acc[1][j]);
    }
    __syncthreads();
  }

  int rowoff = (lane & 16) ? 8 : 0;
  int cl = lane & 15;
#pragma unroll
  for (int i = 0; i < 2; ++i)
#pragma unroll
    for (int j = 0; j < 4; ++j) {
      int mr = m0 + mt + i * 16 + rowoff;
      int nc = n0 + nt + j * 16 + cl;
      float bv = bias[nc];
#pragma unroll
      for (int r = 0; r < 8; ++r)
        Cout[(size_t)(mr + r) * G_DIM + nc] = (half_t)(acc[i][j][r] + bv);
    }
}

// ------------------------------------------------------------------ GRU scan
// 8 blocks x 16 batch rows. h (16x512 f16) lives in LDS as the WMMA A matrix.
// Per step, per gate g in {r,z,n}, per 256-col subchunk: hg = h @ w_hh[:,cols]
// with B fragments loaded straight from transposed w_hh (L2-resident); no
// barriers inside the K loop. Gate math couples (r,z,n) via an LDS hg spill.
__global__ __launch_bounds__(256) void k_gru(const half_t* __restrict__ xg,
                                             const half_t* __restrict__ whhT,
                                             half_t* __restrict__ hout) {
  __shared__ half_t hs[16 * 512];    // 16 KB h state (f16)
  __shared__ float uf[16 * 256];     // 16 KB hg spill
  __shared__ half_t rbuf[16 * 512];  // 16 KB r gate (then h_new stash)
  __shared__ half_t zbuf[16 * 512];  // 16 KB z gate
  int b0 = blockIdx.x * 16;
  int tid = threadIdx.x, lane = tid & 31, wave = tid >> 5;

  for (int i = tid; i < 16 * 512; i += 256) hs[i] = (half_t)0.f;
  __syncthreads();

  for (int t = 0; t < T_DIM; ++t) {
    for (int g = 0; g < 3; ++g) {
      for (int sc = 0; sc < 2; ++sc) {
        int cb = g * 512 + sc * 256;       // column base in [0,1536)
        int ncol = cb + wave * 32;         // this wave's 2 tiles
        v8f acc0 = vzero8(), acc1 = vzero8();
        const half_t* w0 = whhT + (size_t)ncol * H_DIM;
        const half_t* w1 = whhT + (size_t)(ncol + 16) * H_DIM;
#pragma unroll
        for (int ksx = 0; ksx < 16; ++ksx) {
          int k0 = ksx * 32;
          v16h af = load_frag(&hs[k0], 512, lane);
          v16h bf0 = load_frag(w0 + k0, H_DIM, lane);
          v16h bf1 = load_frag(w1 + k0, H_DIM, lane);
          acc0 = wmma_f16(af, bf0, acc0);
          acc1 = wmma_f16(af, bf1, acc1);
        }
        {  // spill hg subchunk (16x256 f32)
          int rowoff = (lane & 16) ? 8 : 0;
          int cl = lane & 15;
          int c0 = wave * 32 + cl, c1 = c0 + 16;
#pragma unroll
          for (int r = 0; r < 8; ++r) {
            uf[(r + rowoff) * 256 + c0] = acc0[r];
            uf[(r + rowoff) * 256 + c1] = acc1[r];
          }
        }
        __syncthreads();
        // gate math: 4096 elems, 16 contiguous per thread
#pragma unroll
        for (int e = 0; e < 16; ++e) {
          int id = tid * 16 + e;
          int m = id >> 8, c = id & 255;
          int row = (b0 + m) * T_DIM + t;
          float xv = (float)xg[(size_t)row * G_DIM + cb + c];
          float hg = uf[m * 256 + c];
          int hc = sc * 256 + c;
          if (g == 0) {
            rbuf[m * 512 + hc] = (half_t)(1.f / (1.f + __expf(-(xv + hg))));
          } else if (g == 1) {
            zbuf[m * 512 + hc] = (half_t)(1.f / (1.f + __expf(-(xv + hg))));
          } else {
            float rg = (float)rbuf[m * 512 + hc];
            float zg = (float)zbuf[m * 512 + hc];
            float nn = tanhf(xv + rg * hg);
            float ho = (float)hs[m * 512 + hc];
            rbuf[m * 512 + hc] = (half_t)((1.f - zg) * nn + zg * ho);
          }
        }
        __syncthreads();
      }
    }
    for (int i = tid; i < 16 * 512; i += 256) {  // commit h_new
      half_t hv = rbuf[i];
      hs[i] = hv;
      int m = i >> 9, c = i & 511;
      hout[((size_t)(b0 + m) * T_DIM + t) * H_DIM + c] = hv;
    }
    __syncthreads();
  }
}

// --------------------------------------------------- collapsed attention query
__global__ __launch_bounds__(256) void k_qv(const float* __restrict__ q,
                                            const float* __restrict__ wqw,
                                            const float* __restrict__ wqb,
                                            const float* __restrict__ bilw,
                                            const float* __restrict__ wkw,
                                            float* __restrict__ v) {
  __shared__ float qx[D_DIM];
  __shared__ float qw[D_DIM];
  int tid = threadIdx.x;
  {
    float a0 = wqb[tid * 4 + 0], a1 = wqb[tid * 4 + 1];
    float a2 = wqb[tid * 4 + 2], a3 = wqb[tid * 4 + 3];
    for (int i = 0; i < D_DIM; ++i) {
      float qi = q[i];
      const float* w = wqw + (size_t)i * D_DIM + tid * 4;
      a0 += qi * w[0]; a1 += qi * w[1]; a2 += qi * w[2]; a3 += qi * w[3];
    }
    qx[tid * 4 + 0] = a0; qx[tid * 4 + 1] = a1;
    qx[tid * 4 + 2] = a2; qx[tid * 4 + 3] = a3;
  }
  __syncthreads();
  {
    float a0 = 0.f, a1 = 0.f, a2 = 0.f, a3 = 0.f;
    for (int i = 0; i < D_DIM; ++i) {
      float qi = qx[i];
      const float* w = bilw + (size_t)i * D_DIM + tid * 4;
      a0 += qi * w[0]; a1 += qi * w[1]; a2 += qi * w[2]; a3 += qi * w[3];
    }
    qw[tid * 4 + 0] = a0; qw[tid * 4 + 1] = a1;
    qw[tid * 4 + 2] = a2; qw[tid * 4 + 3] = a3;
  }
  __syncthreads();
#pragma unroll
  for (int r = 0; r < 2; ++r) {
    int i = tid * 2 + r;  // i < 512
    float s = 0.f;
    const float* w = wkw + (size_t)i * D_DIM;
    for (int j = 0; j < D_DIM; ++j) s += w[j] * qw[j];
    v[i] = s;
  }
}

// --------------------------------------------------------------- score logits
__global__ __launch_bounds__(256) void k_logits(const half_t* __restrict__ h,
                                                const float* __restrict__ v,
                                                float* __restrict__ lg) {
  int wave = threadIdx.x >> 5, lane = threadIdx.x & 31;
  int bt = blockIdx.x * 8 + wave;
  const half_t* hr = h + (size_t)bt * H_DIM;
  float s = 0.f;
#pragma unroll
  for (int k = lane; k < H_DIM; k += 32) s += (float)hr[k] * v[k];
#pragma unroll
  for (int m = 16; m >= 1; m >>= 1) s += __shfl_xor(s, m, 32);
  if (lane == 0) lg[bt] = s;
}

// ------------------------------------------ softmax + pooled + dense head
__global__ __launch_bounds__(256) void k_softpool(const float* __restrict__ lg,
                                                  const half_t* __restrict__ h,
                                                  const float* __restrict__ dw,
                                                  const float* __restrict__ db,
                                                  float* __restrict__ out) {
  __shared__ float red[256];
  __shared__ float scr[256];
  __shared__ float pl[512];
  int b = blockIdx.x, tid = threadIdx.x;
  float x = lg[b * T_DIM + tid];
  red[tid] = x;
  __syncthreads();
  for (int s = 128; s >= 1; s >>= 1) {
    if (tid < s) red[tid] = fmaxf(red[tid], red[tid + s]);
    __syncthreads();
  }
  float mx = red[0];
  __syncthreads();
  float e = __expf(x - mx);
  red[tid] = e;
  __syncthreads();
  for (int s = 128; s >= 1; s >>= 1) {
    if (tid < s) red[tid] += red[tid + s];
    __syncthreads();
  }
  scr[tid] = e / red[0];
  __syncthreads();
  const half_t* hb = h + (size_t)b * T_DIM * H_DIM;
  for (int k = tid; k < H_DIM; k += 256) {
    float acc = 0.f;
    for (int t = 0; t < T_DIM; ++t) acc += scr[t] * (float)hb[(size_t)t * H_DIM + k];
    pl[k] = acc;
  }
  __syncthreads();
  if (tid < POL_N) {
    float o = db[tid];
    for (int k = 0; k < H_DIM; ++k) o += pl[k] * dw[k * POL_N + tid];
    out[b * POL_N + tid] = o;
  }
}

// ------------------------------------------------------------------- launcher
extern "C" void kernel_launch(void* const* d_in, const int* in_sizes, int n_in,
                              void* d_out, int out_size, void* d_ws,
                              size_t ws_size, hipStream_t stream) {
  (void)in_sizes; (void)n_in; (void)out_size; (void)ws_size;
  const int* text = (const int*)d_in[0];
  const int* aidx = (const int*)d_in[1];
  const float* emb = (const float*)d_in[2];
  const float* wih = (const float*)d_in[3];
  const float* whh = (const float*)d_in[4];
  const float* bgru = (const float*)d_in[5];
  const float* wkw = (const float*)d_in[6];
  // d_in[7] = wk_b: per-batch constant in the logits -> cancels in softmax
  const float* wqw = (const float*)d_in[8];
  const float* wqb = (const float*)d_in[9];
  const float* q = (const float*)d_in[10];
  const float* bilw = (const float*)d_in[11];
  const float* dw = (const float*)d_in[12];
  const float* db = (const float*)d_in[13];
  float* out = (float*)d_out;

  char* ws = (char*)d_ws;
  size_t off = 0;
  auto alloc = [&](size_t bytes) -> char* {
    char* p = ws + off;
    off += (bytes + 255) & ~(size_t)255;
    return p;
  };
  float* asp = (float*)alloc((size_t)B_DIM * E_DIM * 4);
  half_t* wihT = (half_t*)alloc((size_t)I_DIM * G_DIM * 2);  // [1536][1024]
  half_t* whhT = (half_t*)alloc((size_t)H_DIM * G_DIM * 2);  // [1536][512]
  float* v = (float*)alloc((size_t)H_DIM * 4);
  float* lgt = (float*)alloc((size_t)M_DIM * 4);
  half_t* xin = (half_t*)alloc((size_t)M_DIM * I_DIM * 2);   // 64 MB
  half_t* xgh = (half_t*)alloc((size_t)M_DIM * G_DIM * 2);   // 96 MB
  half_t* hout = (half_t*)alloc((size_t)M_DIM * H_DIM * 2);  // 32 MB

  k_aspool<<<B_DIM, 256, 0, stream>>>(aidx, emb, asp);
  k_cvtT<<<(I_DIM / 32) * (G_DIM / 32), 256, 0, stream>>>(wih, wihT, I_DIM, G_DIM);
  k_cvtT<<<(H_DIM / 32) * (G_DIM / 32), 256, 0, stream>>>(whh, whhT, H_DIM, G_DIM);
  k_xin<<<M_DIM, 256, 0, stream>>>(text, emb, asp, xin);
  k_gemm_xg<<<256 * 12, 256, 0, stream>>>(xin, wihT, bgru, xgh);
  k_gru<<<8, 256, 0, stream>>>(xgh, whhT, hout);
  k_qv<<<1, 256, 0, stream>>>(q, wqw, wqb, bilw, wkw, v);
  k_logits<<<M_DIM / 8, 256, 0, stream>>>(hout, v, lgt);
  k_softpool<<<B_DIM, 256, 0, stream>>>(lgt, hout, dw, db, out);
}
